// MSClapAudioSelfAttention_73744588472690
// MI455X (gfx1250) — matrix operand order repaired and checked by reading, not compiled, not run on hardware
//
#include <hip/hip_runtime.h>

typedef _Float16 half_t;
typedef __attribute__((ext_vector_type(16))) _Float16 v16h;
typedef __attribute__((ext_vector_type(8)))  _Float16 v8h;
typedef __attribute__((ext_vector_type(8)))  float    v8f;

#define WH_ 8
#define WW_ 8
#define NTOK 64
#define DIM_ 256
#define HEADS_ 8
#define HDIM 32
#define NWIN 2048
#define MASK_NW_ 64

// LDS row strides (in halves), padded, rows 16B aligned
#define XSTR 264   // 256 + 8
#define VSTR 72    // 64 + 8
#define PSTR 72

// ---------------- fragment loaders ----------------
// A operand (16x32, f16): lane L holds row M = L%16.
// halves 0..7  = K[k0 + 8g .. +8),  halves 8..15 = K[k0 + 16 + 8g .. +8), g = L/16
__device__ __forceinline__ v16h load_fragA(const half_t* base, int rowStart,
                                           int stride, int k0, int lane) {
    int rr = rowStart + (lane & 15);
    int g  = lane >> 4;
    const half_t* p = base + rr * stride + k0 + g * 8;
    v8h lo = *(const v8h*)(p);
    v8h hi = *(const v8h*)(p + 16);
    return __builtin_shufflevector(lo, hi, 0,1,2,3,4,5,6,7,8,9,10,11,12,13,14,15);
}

// B operand (32x16, f16): lane L holds column N = L%16.
// halves 0..7  = K[k0 + 8g .. +8),  halves 8..15 = K[k0 + 16 + 8g .. +8), g = L/16
// Source matrix stored row-major with the WMMA N index as the matrix row index.
__device__ __forceinline__ v16h load_fragB(const half_t* base, int colStart,
                                           int stride, int k0, int lane) {
    int nn = colStart + (lane & 15);
    int g  = lane >> 4;
    const half_t* p = base + nn * stride + k0 + g * 8;
    v8h lo = *(const v8h*)(p);
    v8h hi = *(const v8h*)(p + 16);
    return __builtin_shufflevector(lo, hi, 0,1,2,3,4,5,6,7,8,9,10,11,12,13,14,15);
}

__device__ __forceinline__ v8f wmma_f16(v16h a, v16h b, v8f c) {
    return __builtin_amdgcn_wmma_f32_16x16x32_f16(false, a, false, b,
                                                  (short)0, c, false, false);
}

// ---------------- prep kernels ----------------
__global__ void prep_weights_kernel(const float* __restrict__ wq,
                                    const float* __restrict__ wk,
                                    const float* __restrict__ wv,
                                    half_t* __restrict__ w16) {
    int i = blockIdx.x * 256 + threadIdx.x;   // 0..65535
    w16[i]              = (half_t)wq[i];
    w16[65536 + i]      = (half_t)wk[i];
    w16[2 * 65536 + i]  = (half_t)wv[i];
}

__global__ void prep_bias_kernel(const float* __restrict__ table,
                                 const int* __restrict__ rel,
                                 float* __restrict__ rb) {
    int i = blockIdx.x * 256 + threadIdx.x;   // 0..4095 (i over [64][64])
    int idx = rel[i];
#pragma unroll
    for (int h = 0; h < HEADS_; ++h)
        rb[h * 4096 + i] = table[idx * HEADS_ + h];
}

// ---------------- fused window attention ----------------
__global__ __launch_bounds__(256)
void swin_attn_kernel(const float* __restrict__ hidden,
                      const float* __restrict__ mask,
                      const half_t* __restrict__ w16,
                      const float* __restrict__ bq,
                      const float* __restrict__ bk,
                      const float* __restrict__ bv,
                      const float* __restrict__ relb,
                      float* __restrict__ out) {
    __shared__ __align__(16) half_t Qs[NTOK * XSTR];    // 33792 B
    __shared__ __align__(16) half_t Ks[NTOK * XSTR];    // 33792 B
    __shared__ __align__(16) half_t VTs[DIM_ * VSTR];   // 36864 B
    __shared__ __align__(16) half_t SCR[NTOK * XSTR];   // X then per-wave P strips

    const int blk  = blockIdx.x;
    const int tid  = threadIdx.x;
    const int lane = tid & 31;
    const int wave = tid >> 5;
    const int g    = lane >> 4;
    const int ln   = lane & 15;

    // ---- Phase A: hidden[b] f32 -> f16 in SCR ----
    {
        const float4* h4 = reinterpret_cast<const float4*>(hidden) + (size_t)blk * 4096;
#pragma unroll
        for (int it = 0; it < 16; ++it) {
            int idx = tid + 256 * it;        // 0..4095 float4s
            float4 f = h4[idx];
            int row = idx >> 6;              // 64 float4 per 256-wide row
            int c4  = (idx & 63) << 2;
            half_t* dst = SCR + row * XSTR + c4;
            dst[0] = (half_t)f.x; dst[1] = (half_t)f.y;
            dst[2] = (half_t)f.z; dst[3] = (half_t)f.w;
        }
    }
    __syncthreads();

    // ---- Phase B: Q,K,V = X*W^T + b  (192 tiles of 16x16, K=256) ----
#pragma unroll 1
    for (int t = wave; t < 192; t += 8) {
        int s  = t >> 6;          // 0:Q 1:K 2:V
        int tm = (t >> 4) & 3;    // token tile
        int tn = t & 15;          // dim tile
        const half_t* W    = w16 + s * 65536;
        const float*  bvec = (s == 0) ? bq : (s == 1) ? bk : bv;
        v8f acc = {};
#pragma unroll
        for (int kk = 0; kk < DIM_; kk += 32) {
            v16h a = load_fragA(SCR, tm * 16, XSTR, kk, lane);
            v16h b = load_fragB(W,   tn * 16, DIM_, kk, lane);
            acc = wmma_f16(a, b, acc);
        }
        float bb = bvec[tn * 16 + ln];
        if (s == 2) {
            // store transposed: VT[dim][token], lane writes 8 contiguous tokens
            int n = tn * 16 + ln;
            v8h pk;
#pragma unroll
            for (int r = 0; r < 8; ++r) pk[r] = (half_t)(acc[r] + bb);
            *(v8h*)(VTs + n * VSTR + tm * 16 + g * 8) = pk;
        } else {
            half_t* dst = (s == 0) ? Qs : Ks;
            int n = tn * 16 + ln;
#pragma unroll
            for (int r = 0; r < 8; ++r)
                dst[(tm * 16 + r + 8 * g) * XSTR + n] = (half_t)(acc[r] + bb);
        }
    }
    __syncthreads();   // X (in SCR) dead after this point; SCR reused for P

    // ---- Phase C/D: wave w handles head w ----
    const int h  = wave;
    const int d0 = h * HDIM;
    half_t* Pw = SCR + h * (16 * PSTR);              // 16-row P strip per wave
    const float* rbh = relb + h * 4096;
    const float* mk  = mask + (size_t)(blk & (MASK_NW_ - 1)) * 4096;
    const float scale = 0.17677669529663689f;        // 1/sqrt(32)

#pragma unroll 1
    for (int ti = 0; ti < 4; ++ti) {
        // scores: S[ti*16 .. +16][0..64) = Q_h K_h^T
        v16h aq = load_fragA(Qs, ti * 16, XSTR, d0, lane);
        v8f c[4];
#pragma unroll
        for (int tj = 0; tj < 4; ++tj) {
            v16h bk_ = load_fragB(Ks, tj * 16, XSTR, d0, lane);
            v8f z = {};
            c[tj] = wmma_f16(aq, bk_, z);
        }
        // fused scale + relative-position bias + window mask
#pragma unroll
        for (int tj = 0; tj < 4; ++tj) {
            int j = tj * 16 + ln;
#pragma unroll
            for (int r = 0; r < 8; ++r) {
                int i = ti * 16 + r + 8 * g;
                c[tj][r] = c[tj][r] * scale + rbh[i * 64 + j] + mk[i * 64 + j];
            }
        }
        // row softmax: each 16-lane group holds one row's 16 cols per tile
#pragma unroll
        for (int r = 0; r < 8; ++r) {
            float m = fmaxf(fmaxf(c[0][r], c[1][r]), fmaxf(c[2][r], c[3][r]));
            m = fmaxf(m, __shfl_xor(m, 1));
            m = fmaxf(m, __shfl_xor(m, 2));
            m = fmaxf(m, __shfl_xor(m, 4));
            m = fmaxf(m, __shfl_xor(m, 8));
            float ssum = 0.f;
#pragma unroll
            for (int tj = 0; tj < 4; ++tj) {
                float e = __expf(c[tj][r] - m);
                c[tj][r] = e;
                ssum += e;
            }
            ssum += __shfl_xor(ssum, 1);
            ssum += __shfl_xor(ssum, 2);
            ssum += __shfl_xor(ssum, 4);
            ssum += __shfl_xor(ssum, 8);
            float inv = 1.0f / ssum;
            int rl = r + 8 * g;
#pragma unroll
            for (int tj = 0; tj < 4; ++tj)
                Pw[rl * PSTR + tj * 16 + ln] = (half_t)(c[tj][r] * inv);
        }
        // ctx[ti strip] = P * V  (K over 64 tokens, B operand = VT rows)
        v16h ap0 = load_fragA(Pw, 0, PSTR, 0,  lane);
        v16h ap1 = load_fragA(Pw, 0, PSTR, 32, lane);
#pragma unroll
        for (int tn = 0; tn < 2; ++tn) {
            v8f acc = {};
            v16h b0 = load_fragB(VTs, d0 + tn * 16, VSTR, 0,  lane);
            v16h b1 = load_fragB(VTs, d0 + tn * 16, VSTR, 32, lane);
            acc = wmma_f16(ap0, b0, acc);
            acc = wmma_f16(ap1, b1, acc);
            int n = d0 + tn * 16 + ln;
            float* op = out + (size_t)blk * (NTOK * DIM_) + n;
#pragma unroll
            for (int r = 0; r < 8; ++r)
                op[(ti * 16 + r + 8 * g) * DIM_] = acc[r];
        }
    }
}

extern "C" void kernel_launch(void* const* d_in, const int* in_sizes, int n_in,
                              void* d_out, int out_size, void* d_ws, size_t ws_size,
                              hipStream_t stream) {
    const float* hidden = (const float*)d_in[0];
    const float* mask   = (const float*)d_in[1];
    const float* wq     = (const float*)d_in[2];
    const float* bq     = (const float*)d_in[3];
    const float* wk     = (const float*)d_in[4];
    const float* bk     = (const float*)d_in[5];
    const float* wv     = (const float*)d_in[6];
    const float* bv     = (const float*)d_in[7];
    const float* table  = (const float*)d_in[8];
    const int*   rel    = (const int*)d_in[9];
    float* out = (float*)d_out;

    // workspace: [f16 Wq|Wk|Wv : 3*65536*2 B][f32 rel_bias 8*64*64 : 131072 B]
    half_t* w16  = (half_t*)d_ws;
    float*  relb = (float*)((char*)d_ws + 3 * 65536 * sizeof(half_t));

    prep_weights_kernel<<<256, 256, 0, stream>>>(wq, wk, wv, w16);
    prep_bias_kernel<<<16, 256, 0, stream>>>(table, rel, relb);
    swin_attn_kernel<<<NWIN, 256, 0, stream>>>(hidden, mask, w16,
                                               bq, bk, bv, relb, out);
}
